// ShowAttendTellCore_72052371358101
// MI455X (gfx1250) — matrix operand order repaired and checked by reading, not compiled, not run on hardware
//
#include <hip/hip_runtime.h>
#include <hip/hip_bf16.h>

#define B_SZ   128
#define ATT_N  196
#define IN_ENC 512
#define RNN_   1024
#define ATT_F  2048
#define ATT_HD 512

typedef __attribute__((ext_vector_type(16))) __bf16 v16bf;
typedef __attribute__((ext_vector_type(8)))  __bf16 v8bf;
typedef __attribute__((ext_vector_type(8)))  float  v8f;

// ---------------- fragment loaders ----------------
// A fragment 16x32 bf16 (converted from fp32 row-major A), contiguous k order (B uses the same):
// lane L: m = L&15; kh = (L>>4)*16; elems 0..15 -> K=k0+kh+j
__device__ inline v16bf load_a_frag(const float* __restrict__ p) {
  v8bf lo = __builtin_convertvector(*(const v8f*)(p), v8bf);
  v8bf hi = __builtin_convertvector(*(const v8f*)(p + 8), v8bf);
  return __builtin_shufflevector(lo, hi, 0,1,2,3,4,5,6,7,8,9,10,11,12,13,14,15);
}

// ---------------- GEMM: C[M,N] = A(f32,[M,K]) x Bt(bf16,[N,K])^T ----------------
// Block: 256 threads = 8 waves; wave computes 16 rows x 128 cols (8 fragments).
// K, N compile-time so all tile strides become instruction-immediate offsets.
// MODE 0: out0[row,col] = acc + bias[col]
// MODE 1: atomicAdd(out0[row], sum_col tanh(acc+bias[col]+ex0[b,col])*ex1[col])
// MODE 2: v = tanh(acc + bias[col] + ex0[row,col]); out0=v; out1=v
template<int MODE, int K, int N>
__global__ __launch_bounds__(256) void k_gemm(
    const float* __restrict__ A, const __bf16* __restrict__ Bt,
    const float* __restrict__ bias, const float* __restrict__ ex0,
    const float* __restrict__ ex1, float* __restrict__ out0,
    float* __restrict__ out1)
{
  const int lane  = threadIdx.x & 31;
  const int wave  = threadIdx.x >> 5;
  const int mBase = blockIdx.y * 128 + wave * 16;
  const int nBase = blockIdx.x * 128;

  v8f acc[8];
  #pragma unroll
  for (int t = 0; t < 8; ++t) {
    v8f z = {0.f,0.f,0.f,0.f,0.f,0.f,0.f,0.f};
    acc[t] = z;
  }

  // per-lane base pointers (all tile / k-step displacements are constants)
  const int l15 = lane & 15;
  const float*  pa = A  + (size_t)(mBase + l15) * K + (lane >> 4) * 16;
  const __bf16* pb = Bt + (size_t)(nBase + l15) * K + (lane >> 4) * 16;

  for (int k0 = 0; k0 < K; k0 += 32) {
    v16bf a = load_a_frag(pa + k0);
    #pragma unroll
    for (int t = 0; t < 8; ++t) {
      // B fragment 32x16: lane-half selects K block 0-15/16-31, contiguous 32B
      v16bf b = *(const v16bf*)(pb + (size_t)t * 16 * K + k0);
      acc[t] = __builtin_amdgcn_wmma_f32_16x16x32_bf16(
          false, a, false, b, (short)0, acc[t], false, false);
    }
  }

  // C/D layout: lanes 0-15: row = mBase+i, col = nBase+t*16+lane
  //             lanes 16-31: row = mBase+8+i, col = nBase+t*16+(lane-16)
  const int rlo = (lane < 16) ? 0 : 8;
  const int cc  = l15;

  if (MODE == 0) {
    #pragma unroll
    for (int t = 0; t < 8; ++t) {
      int col = nBase + t * 16 + cc;
      float bi = bias[col];
      #pragma unroll
      for (int i = 0; i < 8; ++i) {
        int row = mBase + rlo + i;
        out0[(size_t)row * N + col] = acc[t][i] + bi;
      }
    }
  } else if (MODE == 1) {
    float s[8] = {0,0,0,0,0,0,0,0};
    #pragma unroll
    for (int t = 0; t < 8; ++t) {
      int col = nBase + t * 16 + cc;
      float bi = bias[col];
      float wA = ex1[col];
      #pragma unroll
      for (int i = 0; i < 8; ++i) {
        int row = mBase + rlo + i;
        int bb  = row / ATT_N;                 // const divisor -> mul/shift
        float ah = ex0[(size_t)bb * ATT_HD + col];
        s[i] += tanhf(acc[t][i] + bi + ah) * wA;
      }
    }
    #pragma unroll
    for (int i = 0; i < 8; ++i) {
      float v = s[i];
      v += __shfl_xor(v, 1, 32);
      v += __shfl_xor(v, 2, 32);
      v += __shfl_xor(v, 4, 32);
      v += __shfl_xor(v, 8, 32);   // per-16-lane-half sum over 16 cols
      if (cc == 0) atomicAdd(&out0[mBase + rlo + i], v);
    }
  } else {
    #pragma unroll
    for (int t = 0; t < 8; ++t) {
      int col = nBase + t * 16 + cc;
      float bi = bias[col];
      #pragma unroll
      for (int i = 0; i < 8; ++i) {
        size_t idx = (size_t)(mBase + rlo + i) * N + col;
        float v = tanhf(acc[t][i] + bi + ex0[idx]);
        out0[idx] = v;
        out1[idx] = v;
      }
    }
  }
}

// ---------------- helpers ----------------
__global__ void k_zero(float* __restrict__ p, int n) {
  int i = blockIdx.x * blockDim.x + threadIdx.x;
  if (i < n) p[i] = 0.f;
}

// in: [K,N] fp32 row-major  ->  out: [N,K] bf16 row-major (transposed)
__global__ void k_convert_t(const float* __restrict__ in, __bf16* __restrict__ out,
                            int K, int N) {
  size_t total = (size_t)K * N;
  for (size_t idx = (size_t)blockIdx.x * blockDim.x + threadIdx.x; idx < total;
       idx += (size_t)gridDim.x * blockDim.x) {
    size_t n = idx / K, k = idx % K;
    out[idx] = (__bf16)in[k * (size_t)N + n];
  }
}

__global__ __launch_bounds__(256) void k_softmax(const float* __restrict__ scores,
                                                 float* __restrict__ alpha) {
  __shared__ float red[256];
  const int b = blockIdx.x, t = threadIdx.x;
  float v = (t < ATT_N) ? scores[b * ATT_N + t] : -3.0e38f;
  red[t] = v; __syncthreads();
  for (int s = 128; s > 0; s >>= 1) { if (t < s) red[t] = fmaxf(red[t], red[t + s]); __syncthreads(); }
  float mx = red[0]; __syncthreads();
  float e = (t < ATT_N) ? expf(v - mx) : 0.f;
  red[t] = e; __syncthreads();
  for (int s = 128; s > 0; s >>= 1) { if (t < s) red[t] += red[t + s]; __syncthreads(); }
  float inv = 1.f / red[0];
  if (t < ATT_N) alpha[b * ATT_N + t] = e * inv;
}

// att_res[b,f] = sum_n alpha[b,n] * att_feats[b,n,f]   (memory-bound pass)
__global__ __launch_bounds__(256) void k_att_res(const float* __restrict__ alpha,
                                                 const float* __restrict__ att_feats,
                                                 float* __restrict__ att_res) {
  const int b = blockIdx.y;
  const int f = blockIdx.x * 256 + threadIdx.x;
  const float* base = att_feats + ((size_t)b * ATT_N) * ATT_F + f;
  const float* al = alpha + b * ATT_N;
  float acc = 0.f;
  for (int n = 0; n < ATT_N; ++n) acc += al[n] * base[(size_t)n * ATT_F];
  att_res[(size_t)b * ATT_F + f] = acc;
}

// ---------------- launch ----------------
extern "C" void kernel_launch(void* const* d_in, const int* in_sizes, int n_in,
                              void* d_out, int out_size, void* d_ws, size_t ws_size,
                              hipStream_t stream) {
  (void)in_sizes; (void)n_in; (void)out_size; (void)ws_size;

  const float* xt        = (const float*)d_in[0];
  const float* att_feats = (const float*)d_in[2];
  const float* state_h   = (const float*)d_in[3];   // [1,B,RNN] contiguous
  const float* state_c   = (const float*)d_in[4];
  const float* W_ctx2att = (const float*)d_in[5];
  const float* b_ctx2att = (const float*)d_in[6];
  const float* W_h2att   = (const float*)d_in[7];
  const float* b_h2att   = (const float*)d_in[8];
  const float* W_alpha   = (const float*)d_in[9];
  const float* W_h2rnn   = (const float*)d_in[11];
  const float* b_h2rnn   = (const float*)d_in[12];
  const float* W_att2rnn = (const float*)d_in[13];
  const float* b_att2rnn = (const float*)d_in[14];

  float* out_alpha = (float*)d_out;                 // [B, ATT_N]
  float* out_h     = out_alpha + B_SZ * ATT_N;      // [B, RNN]
  float* out_newh  = out_h + B_SZ * RNN_;           // [1, B, RNN]
  float* out_c     = out_newh + B_SZ * RNN_;        // [1, B, RNN]

  char* ws = (char*)d_ws;
  size_t off = 0;
  auto alloc = [&](size_t bytes) -> char* {
    char* p = ws + off;
    off += (bytes + 255) & ~(size_t)255;
    return p;
  };
  float*  scores  = (float*)alloc((size_t)B_SZ * ATT_N * 4);
  float*  att_h   = (float*)alloc((size_t)B_SZ * ATT_HD * 4);
  float*  att_res = (float*)alloc((size_t)B_SZ * ATT_F * 4);
  float*  tmp     = (float*)alloc((size_t)B_SZ * RNN_ * 4);
  __bf16* Wt_ctx  = (__bf16*)alloc((size_t)ATT_HD * ATT_F * 2);
  __bf16* Wt_h2a  = (__bf16*)alloc((size_t)ATT_HD * RNN_ * 2);
  __bf16* Wt_h2r  = (__bf16*)alloc((size_t)RNN_ * IN_ENC * 2);
  __bf16* Wt_a2r  = (__bf16*)alloc((size_t)RNN_ * ATT_F * 2);

  dim3 blk(256);

  // init scores = 0 (b_alpha cancels in softmax)
  k_zero<<<(B_SZ * ATT_N + 255) / 256, blk, 0, stream>>>(scores, B_SZ * ATT_N);

  // weight conversions (fp32 [K,N] -> bf16 [N,K])
  k_convert_t<<<2048, blk, 0, stream>>>(W_ctx2att, Wt_ctx, ATT_F, ATT_HD);
  k_convert_t<<<1024, blk, 0, stream>>>(W_h2att,   Wt_h2a, RNN_,  ATT_HD);
  k_convert_t<<<1024, blk, 0, stream>>>(W_h2rnn,   Wt_h2r, IN_ENC, RNN_);
  k_convert_t<<<2048, blk, 0, stream>>>(W_att2rnn, Wt_a2r, ATT_F, RNN_);

  // att_h = h @ W_h2att + b_h2att              [128,512]
  k_gemm<0, RNN_, ATT_HD><<<dim3(ATT_HD / 128, B_SZ / 128), blk, 0, stream>>>(
      state_h, Wt_h2a, b_h2att, nullptr, nullptr, att_h, nullptr);

  // scores[b,n] = sum_h tanh(att_feats@W_ctx2att + b + att_h) * W_alpha  (fused)
  k_gemm<1, ATT_F, ATT_HD><<<dim3(ATT_HD / 128, (B_SZ * ATT_N) / 128), blk, 0, stream>>>(
      att_feats, Wt_ctx, b_ctx2att, att_h, W_alpha, scores, nullptr);

  // alpha = softmax(scores)
  k_softmax<<<B_SZ, blk, 0, stream>>>(scores, out_alpha);

  // att_res = alpha @ att_feats
  k_att_res<<<dim3(ATT_F / 256, B_SZ), blk, 0, stream>>>(out_alpha, att_feats, att_res);

  // tmp = xt @ W_h2rnn + b_h2rnn
  k_gemm<0, IN_ENC, RNN_><<<dim3(RNN_ / 128, 1), blk, 0, stream>>>(
      xt, Wt_h2r, b_h2rnn, nullptr, nullptr, tmp, nullptr);

  // output = tanh(tmp + att_res @ W_att2rnn + b_att2rnn) ; new_h = output
  k_gemm<2, ATT_F, RNN_><<<dim3(RNN_ / 128, 1), blk, 0, stream>>>(
      att_res, Wt_a2r, b_att2rnn, tmp, nullptr, out_h, out_newh);

  // state_c passthrough
  hipMemcpyAsync(out_c, state_c, (size_t)B_SZ * RNN_ * sizeof(float),
                 hipMemcpyDeviceToDevice, stream);
}
